// ForgettingAttention_32839319945670
// MI455X (gfx1250) — compile-verified, matrix operand order checked by reading, not run on hardware
//
#include <hip/hip_runtime.h>
#include <hip/hip_bf16.h>
#include <stdint.h>
#include <stddef.h>

// Problem constants (from reference)
#define B_   2
#define S_   2048
#define HID_ 1024
#define H_   16
#define D_   64
#define M_   (B_*S_)     // 4096 rows
#define SCALE_ 0.125f    // 64^-0.5

typedef __attribute__((ext_vector_type(16))) __bf16       v16bf;
typedef __attribute__((ext_vector_type(8)))  float        v8f;
typedef __attribute__((ext_vector_type(4)))  unsigned int u32x4;

union AF  { u32x4 u[2]; v16bf v; };          // 32 bytes: one 16-bit WMMA A/B fragment per lane
union US8 { u32x4 u4; unsigned short s[8]; };

__device__ __forceinline__ unsigned short f2bf(float f) {
  unsigned int u = __float_as_uint(f);
  u += 0x7FFFu + ((u >> 16) & 1u);           // round-to-nearest-even
  return (unsigned short)(u >> 16);
}

__device__ __forceinline__ v8f wmma_bf16(v16bf a, v16bf b, v8f c) {
  // D = A(16x32) * B(32x16) + C, f32 accumulate
  return __builtin_amdgcn_wmma_f32_16x16x32_bf16(false, a, false, b, (short)0, c, false, false);
}

// 16-bit fragment load: lane owns two contiguous 8-element chunks at {ks, ks+16}
// (ks = (lane>>4)*8), matching the ISA 16-bit A/B matrix VGPR layout.
__device__ __forceinline__ v16bf load_frag(const unsigned short* rowbase, int ks) {
  AF f;
  f.u[0] = *reinterpret_cast<const u32x4*>(rowbase + ks);
  f.u[1] = *reinterpret_cast<const u32x4*>(rowbase + ks + 16);
  return f.v;
}

// CDNA5 async global->LDS copy (16B per lane), tracked by ASYNCcnt.
// ldsaddr: wave-relative LDS byte offset (low 32 bits of the generic pointer:
// the shared aperture occupies the high 32 bits only on CDNA5).
__device__ __forceinline__ void async_copy_b128(unsigned ldsaddr, const void* gptr) {
  unsigned long long ga = (unsigned long long)gptr;
  asm volatile("global_load_async_to_lds_b128 %0, %1, off"
               :: "v"(ldsaddr), "v"(ga) : "memory");
}
__device__ __forceinline__ void async_wait0() {
  asm volatile("s_wait_asynccnt 0x0" ::: "memory");
}
__device__ __forceinline__ unsigned lds_off(const void* p) {
  return (unsigned)(size_t)p;
}

__device__ __forceinline__ void store_val(unsigned short* p, size_t i, float v) { p[i] = f2bf(v); }
__device__ __forceinline__ void store_val(float* p,          size_t i, float v) { p[i] = v; }

// ---------------------------------------------------------------------------
// fp32 -> bf16 conversion
// ---------------------------------------------------------------------------
__global__ __launch_bounds__(256) void cvt_kernel(const float* __restrict__ in,
                                                  unsigned short* __restrict__ out, int n) {
  int i = blockIdx.x * 256 + threadIdx.x;
  if (i < n) out[i] = f2bf(in[i]);
}

// ---------------------------------------------------------------------------
// C[M,N] = A[M,K](bf16) * Bw[N,K](bf16)^T  (torch-Linear style).
// Block tile 128x128, BK=32, 8 waves (4x2); each wave computes 32x64 via
// 8 WMMA accumulators. Tiles staged with GLOBAL_LOAD_ASYNC_TO_LDS_B128.
// HEADED: scatter output as [B,H,S,D] (head split of columns) for attention.
// ---------------------------------------------------------------------------
template<bool HEADED, typename OUT>
__global__ __launch_bounds__(256) void gemm_nt(const unsigned short* __restrict__ A,
                                               const unsigned short* __restrict__ Bw,
                                               OUT* __restrict__ C, int M, int N, int K) {
  constexpr int BM = 128, BN = 128, BK = 32, AP = 40; // AP*2 = 80 B pitch, 16B aligned
  __shared__ unsigned short As[BM * AP];
  __shared__ unsigned short Bs[BN * AP];

  const int tid   = threadIdx.x;
  const int rbase = blockIdx.x * BM;
  const int nbase = blockIdx.y * BN;
  const int wave  = tid >> 5, lane = tid & 31;
  const int wm    = wave >> 1, wn  = wave & 1;   // 4 x 2 wave grid
  const int hi    = lane >> 4, lo  = lane & 15;
  const int ks    = hi * 8;

  const v8f zf = {0.f,0.f,0.f,0.f,0.f,0.f,0.f,0.f};
  v8f acc[2][4];
  #pragma unroll
  for (int mt = 0; mt < 2; ++mt)
    #pragma unroll
    for (int nt = 0; nt < 4; ++nt) acc[mt][nt] = zf;

  for (int kt = 0; kt < K; kt += BK) {
    // async stage: 128 rows x 4 chunks of 16B each for A and B (2+2 per thread)
    #pragma unroll
    for (int c = tid; c < BM * (BK/8); c += 256) {
      int row = c >> 2, kc = c & 3;
      async_copy_b128(lds_off(&As[row*AP + kc*8]),
                      A  + (size_t)(rbase + row)*K + kt + kc*8);
      async_copy_b128(lds_off(&Bs[row*AP + kc*8]),
                      Bw + (size_t)(nbase + row)*K + kt + kc*8);
    }
    async_wait0();
    __syncthreads();

    v16bf af[2], bfr[4];
    af[0] = load_frag(&As[(wm*32      + lo)*AP], ks);
    af[1] = load_frag(&As[(wm*32 + 16 + lo)*AP], ks);
    #pragma unroll
    for (int nt = 0; nt < 4; ++nt)
      bfr[nt] = load_frag(&Bs[(wn*64 + nt*16 + lo)*AP], ks);

    #pragma unroll
    for (int mt = 0; mt < 2; ++mt)
      #pragma unroll
      for (int nt = 0; nt < 4; ++nt)
        acc[mt][nt] = wmma_bf16(af[mt], bfr[nt], acc[mt][nt]);
    __syncthreads();
  }

  #pragma unroll
  for (int mt = 0; mt < 2; ++mt)
    #pragma unroll
    for (int nt = 0; nt < 4; ++nt)
      #pragma unroll
      for (int r = 0; r < 8; ++r) {
        const int row_g = rbase + wm*32 + mt*16 + r + 8*hi;
        const int col_g = nbase + wn*64 + nt*16 + lo;
        size_t idx;
        if (HEADED) { // [B,H,S,D]: b=row>>11, s=row&2047, h=col>>6, d=col&63
          idx = (((size_t)(row_g >> 11) * H_ + (col_g >> 6)) * S_ + (row_g & (S_-1))) * D_
              + (col_g & (D_-1));
        } else {
          idx = (size_t)row_g * N + col_g;
        }
        store_val(C, idx, acc[mt][nt][r]);
      }
}

// ---------------------------------------------------------------------------
// Forget-gate logits: flog[b,h,s] = log(sigmoid(x . Wf[h] + bf[h]) + 1e-6)
// ---------------------------------------------------------------------------
__global__ __launch_bounds__(256) void fgate_kernel(const float* __restrict__ x,
                                                    const float* __restrict__ Wf,
                                                    const float* __restrict__ bfv,
                                                    float* __restrict__ flog) {
  const int t   = blockIdx.x * 256 + threadIdx.x;   // 65536 = 4096 rows * 16 heads
  const int row = t >> 4;
  const int h   = t & 15;
  const float* xr = x  + (size_t)row * HID_;
  const float* wr = Wf + (size_t)h   * HID_;
  float acc = 0.f;
  for (int i = 0; i < HID_; ++i) acc += xr[i] * wr[i];
  acc += bfv[h];
  const float sig = 1.f / (1.f + __expf(-acc));
  const int b = row >> 11, s = row & (S_-1);
  flog[((size_t)b * H_ + h) * S_ + s] = __logf(sig + 1e-6f);
}

__global__ void cumsum_kernel(const float* __restrict__ flog, float* __restrict__ cum) {
  const int bh = blockIdx.x * blockDim.x + threadIdx.x;
  if (bh >= B_ * H_) return;
  const float* in = flog + (size_t)bh * S_;
  float* out      = cum  + (size_t)bh * S_;
  float a = 0.f;
  for (int s = 0; s < S_; ++s) { a += in[s]; out[s] = a; }
}

// ---------------------------------------------------------------------------
// Flash attention with decay bias. One block per (64 q rows, b*h).
// 4 waves x 16 q rows; key blocks of 32; online softmax; WMMA for QK^T and PV.
// K tile staged via async global->LDS; V transposed through VGPRs.
// ---------------------------------------------------------------------------
__global__ __launch_bounds__(128) void fa_kernel(const unsigned short* __restrict__ qg,
                                                 const unsigned short* __restrict__ kg,
                                                 const unsigned short* __restrict__ vg,
                                                 const float* __restrict__ cum,
                                                 unsigned short* __restrict__ og) {
  constexpr int KP = 72, VP = 40, PP = 40;  // LDS pitches (ushorts), 16B-aligned rows
  __shared__ unsigned short Ks[32 * KP];    // K tile  [32 keys][64 d]
  __shared__ unsigned short Vt[D_ * VP];    // V tile transposed [64 d][32 keys]
  __shared__ unsigned short Ps[4 * 16 * PP];// per-wave P tiles [16][32]
  __shared__ float cqs[64];
  __shared__ float cks[32];

  const int tid  = threadIdx.x;
  const int wave = tid >> 5, lane = tid & 31;
  const int hi   = lane >> 4, lo  = lane & 15;
  const int ks   = hi * 8;
  const int qb0  = blockIdx.x * 64;
  const int bh   = blockIdx.y;
  const int b    = bh >> 4, h = bh & 15;

  const unsigned short* qbh = qg + (size_t)bh * S_ * D_;
  const unsigned short* kbh = kg + (size_t)bh * S_ * D_;
  const unsigned short* vbh = vg + (size_t)bh * S_ * D_;
  const float* cumbh        = cum + (size_t)bh * S_;

  // Q fragments: row m = lo of this wave's 16-row tile, d split into two K=32 chunks
  const int qrow = qb0 + wave*16 + lo;
  v16bf aQ0 = load_frag(qbh + (size_t)qrow * D_,       ks);
  v16bf aQ1 = load_frag(qbh + (size_t)qrow * D_ + 32,  ks);

  if (tid < 64) cqs[tid] = cumbh[qb0 + tid];
  __syncthreads();
  float cq[8];
  #pragma unroll
  for (int r = 0; r < 8; ++r) cq[r] = cqs[wave*16 + r + 8*hi];

  const v8f zf = {0.f,0.f,0.f,0.f,0.f,0.f,0.f,0.f};
  float mst[8], lst[8];
  v8f o[4];
  #pragma unroll
  for (int r = 0; r < 8; ++r) { mst[r] = -1e30f; lst[r] = 0.f; }
  #pragma unroll
  for (int nt = 0; nt < 4; ++nt) o[nt] = zf;

  const int nkb = qb0/32 + 2;     // causal: only key blocks touching rows <= qb0+63
  for (int jb = 0; jb < nkb; ++jb) {
    const int kb = jb * 32;
    __syncthreads();              // previous iteration's LDS reads done
    // K tile: async global->LDS. V tile: through VGPRs (transposed on store).
    #pragma unroll
    for (int c = tid; c < 256; c += 128) {
      int row = c >> 3, kc = c & 7;
      async_copy_b128(lds_off(&Ks[row*KP + kc*8]),
                      kbh + (size_t)(kb + row)*D_ + kc*8);
      US8 vv; vv.u4 = *reinterpret_cast<const u32x4*>(vbh + (size_t)(kb + row)*D_ + kc*8);
      #pragma unroll
      for (int j = 0; j < 8; ++j) Vt[(kc*8 + j)*VP + row] = vv.s[j];
    }
    if (tid < 32) cks[tid] = cumbh[kb + tid];
    if (jb + 1 < nkb)   // hint next K tile toward the caches
      __builtin_prefetch(kbh + (size_t)(kb + 32)*D_ + (tid & 31)*D_, 0, 0);
    async_wait0();
    __syncthreads();

    // scores: two 16x16 tiles (keys kb+0..15, kb+16..31), K-dim=64 -> 2 WMMA each
    v8f s0 = zf, s1 = zf;
    s0 = wmma_bf16(aQ0, load_frag(&Ks[lo*KP],            ks), s0);
    s0 = wmma_bf16(aQ1, load_frag(&Ks[lo*KP + 32],       ks), s0);
    s1 = wmma_bf16(aQ0, load_frag(&Ks[(16 + lo)*KP],     ks), s1);
    s1 = wmma_bf16(aQ1, load_frag(&Ks[(16 + lo)*KP + 32],ks), s1);

    const float ck0 = cks[lo], ck1 = cks[16 + lo];
    unsigned short* Pw = &Ps[wave * 16 * PP];
    #pragma unroll
    for (int r = 0; r < 8; ++r) {
      const int row_g = qb0 + wave*16 + r + 8*hi;
      const int c0 = kb + lo, c1 = kb + 16 + lo;
      float x0 = (c0 <= row_g) ? (s0[r]*SCALE_ + cq[r] - ck0) : -1e30f;
      float x1 = (c1 <= row_g) ? (s1[r]*SCALE_ + cq[r] - ck1) : -1e30f;
      float mx = fmaxf(x0, x1);
      #pragma unroll
      for (int off = 1; off < 16; off <<= 1) mx = fmaxf(mx, __shfl_xor(mx, off, 16));
      const float mnew  = fmaxf(mst[r], mx);
      const float alpha = __expf(mst[r] - mnew);
      const float p0 = __expf(x0 - mnew);
      const float p1 = __expf(x1 - mnew);
      float rs = p0 + p1;
      #pragma unroll
      for (int off = 1; off < 16; off <<= 1) rs += __shfl_xor(rs, off, 16);
      lst[r] = lst[r] * alpha + rs;
      mst[r] = mnew;
      #pragma unroll
      for (int nt = 0; nt < 4; ++nt) o[nt][r] = o[nt][r] * alpha;
      Pw[(r + 8*hi)*PP + lo]      = f2bf(p0);
      Pw[(r + 8*hi)*PP + 16 + lo] = f2bf(p1);
    }
    __syncthreads();               // P visible to own wave's fragment loads

    v16bf aP = load_frag(&Pw[lo*PP], ks);
    #pragma unroll
    for (int nt = 0; nt < 4; ++nt)
      o[nt] = wmma_bf16(aP, load_frag(&Vt[(nt*16 + lo)*VP], ks), o[nt]);
  }

  // epilogue: normalize and store as bf16 [B,S,HID] for the output projection
  #pragma unroll
  for (int nt = 0; nt < 4; ++nt)
    #pragma unroll
    for (int r = 0; r < 8; ++r) {
      const int srow = qb0 + wave*16 + r + 8*hi;
      const int col  = h*D_ + nt*16 + lo;
      og[((size_t)b*S_ + srow)*HID_ + col] = f2bf(o[nt][r] / lst[r]);
    }
}

// ---------------------------------------------------------------------------
extern "C" void kernel_launch(void* const* d_in, const int* in_sizes, int n_in,
                              void* d_out, int out_size, void* d_ws, size_t ws_size,
                              hipStream_t stream) {
  (void)in_sizes; (void)n_in; (void)out_size; (void)ws_size;
  const float* x  = (const float*)d_in[0];
  const float* Wq = (const float*)d_in[1];
  const float* Wk = (const float*)d_in[2];
  const float* Wv = (const float*)d_in[3];
  const float* Wf = (const float*)d_in[4];
  const float* bf = (const float*)d_in[5];
  const float* Wo = (const float*)d_in[6];
  float* out = (float*)d_out;

  // workspace partition (~51 MB)
  char* w = (char*)d_ws;
  auto alloc = [&](size_t bytes) { char* p = w; w += (bytes + 255) & ~(size_t)255; return p; };
  unsigned short* xb  = (unsigned short*)alloc((size_t)M_ * HID_ * 2);
  unsigned short* wqb = (unsigned short*)alloc((size_t)HID_ * HID_ * 2);
  unsigned short* wkb = (unsigned short*)alloc((size_t)HID_ * HID_ * 2);
  unsigned short* wvb = (unsigned short*)alloc((size_t)HID_ * HID_ * 2);
  unsigned short* wob = (unsigned short*)alloc((size_t)HID_ * HID_ * 2);
  unsigned short* qb  = (unsigned short*)alloc((size_t)M_ * HID_ * 2);
  unsigned short* kb  = (unsigned short*)alloc((size_t)M_ * HID_ * 2);
  unsigned short* vb  = (unsigned short*)alloc((size_t)M_ * HID_ * 2);
  unsigned short* aob = (unsigned short*)alloc((size_t)M_ * HID_ * 2);
  float* flog = (float*)alloc((size_t)B_ * H_ * S_ * 4);
  float* cum  = (float*)alloc((size_t)B_ * H_ * S_ * 4);

  const int nx = M_ * HID_, nw = HID_ * HID_;
  cvt_kernel<<<(nx + 255)/256, 256, 0, stream>>>(x,  xb,  nx);
  cvt_kernel<<<(nw + 255)/256, 256, 0, stream>>>(Wq, wqb, nw);
  cvt_kernel<<<(nw + 255)/256, 256, 0, stream>>>(Wk, wkb, nw);
  cvt_kernel<<<(nw + 255)/256, 256, 0, stream>>>(Wv, wvb, nw);
  cvt_kernel<<<(nw + 255)/256, 256, 0, stream>>>(Wo, wob, nw);

  dim3 gg(M_/128, HID_/128);
  gemm_nt<true,  unsigned short><<<gg, 256, 0, stream>>>(xb, wqb, qb, M_, HID_, HID_);
  gemm_nt<true,  unsigned short><<<gg, 256, 0, stream>>>(xb, wkb, kb, M_, HID_, HID_);
  gemm_nt<true,  unsigned short><<<gg, 256, 0, stream>>>(xb, wvb, vb, M_, HID_, HID_);

  fgate_kernel<<<(M_*H_)/256, 256, 0, stream>>>(x, Wf, bf, flog);
  cumsum_kernel<<<1, 32, 0, stream>>>(flog, cum);

  dim3 ga(S_/64, B_*H_);
  fa_kernel<<<ga, 128, 0, stream>>>(qb, kb, vb, cum, aob);

  gemm_nt<false, float><<<gg, 256, 0, stream>>>(aob, wob, out, M_, HID_, HID_);
}
